// MultiBoxLoss_35974646071653
// MI455X (gfx1250) — compile-verified, matrix operand order checked by reading, not run on hardware
//
#include <hip/hip_runtime.h>

// Problem constants (from reference setup_inputs)
#define P_REAL 24564
#define P_PAD  24576          // 3 * 8192, zero-padded ce buffer
#define B_SZ   32
#define C_CLS  21
#define N_OBJ  24
#define TPB    256
#define BLKS_PER_B (P_PAD / TPB)     // 96
#define NBLK2  (B_SZ * BLKS_PER_B)   // 3072

typedef int v8i __attribute__((ext_vector_type(8)));
typedef int vs4i __attribute__((vector_size(16)));   // matches builtin param type

#if defined(__gfx1250__) && __has_builtin(__builtin_amdgcn_global_load_async_to_lds_b128) && \
    __has_builtin(__builtin_amdgcn_s_wait_asynccnt)
#define ASYNC_LDS_OK 1
#endif

__device__ __forceinline__ float iou_fn(float ax0, float ay0, float ax1, float ay1,
                                        float bx0, float by0, float bx1, float by1) {
    float lx = fmaxf(ax0, bx0), ly = fmaxf(ay0, by0);
    float rx = fminf(ax1, bx1), ry = fminf(ay1, by1);
    float w = fmaxf(rx - lx, 0.f), h = fmaxf(ry - ly, 0.f);
    float inter = w * h;
    float aa = (ax1 - ax0) * (ay1 - ay0);
    float ab = (bx1 - bx0) * (by1 - by0);
    return inter / (aa + ab - inter);
}

__global__ void k_init(int* __restrict__ n_pos) {
    if (threadIdx.x < B_SZ) n_pos[threadIdx.x] = 0;
}

// Stage 1: per (batch, object) argmax over priors of IoU (force-match index).
// argmax-first tie semantics: strictly-greater, or equal with lower prior idx.
__global__ void k_stage1(const float* __restrict__ gt_boxes,
                         const float* __restrict__ anc,
                         int* __restrict__ obj_idx) {
    int b = blockIdx.x, tid = threadIdx.x;
    __shared__ float4 sbox[N_OBJ];
    __shared__ float  sbest[N_OBJ * TPB];
    __shared__ int    sbidx[N_OBJ * TPB];
    if (tid < N_OBJ) sbox[tid] = ((const float4*)gt_boxes)[b * N_OBJ + tid];
    __syncthreads();

    float best[N_OBJ];
    int   bidx[N_OBJ];
#pragma unroll
    for (int i = 0; i < N_OBJ; i++) { best[i] = -1.f; bidx[i] = 0x7fffffff; }

    for (int p = tid; p < P_REAL; p += TPB) {
        float4 a = ((const float4*)anc)[p];
        float ax0 = a.x - a.z * 0.5f, ay0 = a.y - a.w * 0.5f;
        float ax1 = a.x + a.z * 0.5f, ay1 = a.y + a.w * 0.5f;
#pragma unroll
        for (int i = 0; i < N_OBJ; i++) {
            float4 bb = sbox[i];
            float v = iou_fn(bb.x, bb.y, bb.z, bb.w, ax0, ay0, ax1, ay1);
            if (v > best[i] || (v == best[i] && p < bidx[i])) { best[i] = v; bidx[i] = p; }
        }
    }
#pragma unroll
    for (int i = 0; i < N_OBJ; i++) { sbest[i * TPB + tid] = best[i]; sbidx[i * TPB + tid] = bidx[i]; }
    __syncthreads();
    if (tid < N_OBJ) {
        float bv = -1.f; int bi = 0x7fffffff;
        for (int t = 0; t < TPB; t++) {
            float v = sbest[tid * TPB + t]; int ix = sbidx[tid * TPB + t];
            if (v > bv || (v == bv && ix < bi)) { bv = v; bi = ix; }
        }
        obj_idx[b * N_OBJ + tid] = bi;
    }
}

// Stage 2: fused main streaming pass (the HBM-bound 80 MB). Per prior:
// matching (with force-match override), CE via log-softmax from an LDS-staged
// coalesced pred_cls tile (async-to-LDS b128 when available), L1 loc loss,
// ce_neg scratch write, deterministic block reductions.
__global__ void k_stage2(const float* __restrict__ pred_cls,
                         const float* __restrict__ pred_loc,
                         const float* __restrict__ gt_boxes,
                         const int*   __restrict__ gt_labels,
                         const float* __restrict__ anc,
                         const int*   __restrict__ obj_idx,
                         float*  __restrict__ ce_out,
                         int*    __restrict__ n_pos,
                         double* __restrict__ pconf,
                         double* __restrict__ ploc) {
    int b   = blockIdx.x / BLKS_PER_B;
    int blk = blockIdx.x % BLKS_PER_B;
    int tid = threadIdx.x;
    int p   = blk * TPB + tid;

    __shared__ __align__(16) float s_cls[TPB * C_CLS];
    __shared__ float4 sbox[N_OBJ];
    __shared__ int    slab[N_OBJ];
    __shared__ int    sobj[N_OBJ];
    __shared__ double rs[TPB], rl[TPB];
    __shared__ int    rc[TPB];

    if (tid < N_OBJ) {
        sbox[tid] = ((const float4*)gt_boxes)[b * N_OBJ + tid];
        slab[tid] = gt_labels[b * N_OBJ + tid];
        sobj[tid] = obj_idx[b * N_OBJ + tid];
    }
    int row0 = blk * TPB;
    int rowcnt = P_REAL - row0;
    if (rowcnt > TPB) rowcnt = TPB;            // 256 or 244: rowcnt*21 % 4 == 0
    const float* src = pred_cls + ((size_t)b * P_REAL + row0) * C_CLS;
    int nb = (rowcnt * C_CLS) >> 2;            // number of 16-byte chunks
#ifdef ASYNC_LDS_OK
    for (int i = tid; i < nb; i += TPB) {
        __builtin_amdgcn_global_load_async_to_lds_b128(
            (__attribute__((address_space(1))) vs4i*)(uintptr_t)(src + i * 4),
            (__attribute__((address_space(3))) vs4i*)(uintptr_t)(&s_cls[i * 4]),
            0, 0);
    }
    __builtin_amdgcn_s_wait_asynccnt(0);
#else
    for (int i = tid; i < nb; i += TPB)
        *(float4*)(&s_cls[i * 4]) = *(const float4*)(src + i * 4);
#endif
    __syncthreads();

    double my_conf = 0.0, my_loc = 0.0;
    int my_pos = 0;
    float ce_neg = 0.f;

    if (p < P_REAL) {
        float4 a = ((const float4*)anc)[p];
        float ax0 = a.x - a.z * 0.5f, ay0 = a.y - a.w * 0.5f;
        float ax1 = a.x + a.z * 0.5f, ay1 = a.y + a.w * 0.5f;
        float bestv = -1.f; int besti = 0;
#pragma unroll
        for (int i = 0; i < N_OBJ; i++) {
            float4 bb = sbox[i];
            float v = (sobj[i] == p) ? 1.0f
                                     : iou_fn(bb.x, bb.y, bb.z, bb.w, ax0, ay0, ax1, ay1);
            if (v > bestv) { bestv = v; besti = i; }   // first-max like jnp.argmax
        }
        bool pos = (bestv >= 0.5f);
        int tc = pos ? (slab[besti] + 1) : 0;

        const float* xr = &s_cls[tid * C_CLS];
        float m = xr[0];
#pragma unroll
        for (int c = 1; c < C_CLS; c++) m = fmaxf(m, xr[c]);
        float se = 0.f;
#pragma unroll
        for (int c = 0; c < C_CLS; c++) se += expf(xr[c] - m);
        float ce = logf(se) + m - xr[tc];

        if (pos) {
            my_pos = 1;
            my_conf = (double)ce;
            float4 bb = sbox[besti];
            float bcx = (bb.x + bb.z) * 0.5f, bcy = (bb.y + bb.w) * 0.5f;
            float bw = bb.z - bb.x, bh = bb.w - bb.y;
            float t0 = (bcx - a.x) * 10.f / a.z;
            float t1 = (bcy - a.y) * 10.f / a.w;
            float t2 = logf(bw / a.z) * 5.f;
            float t3 = logf(bh / a.w) * 5.f;
            float4 pl = ((const float4*)pred_loc)[(size_t)b * P_REAL + p];
            my_loc = (double)(fabsf(pl.x - t0) + fabsf(pl.y - t1) +
                              fabsf(pl.z - t2) + fabsf(pl.w - t3));
        } else {
            ce_neg = ce;
        }
    }
    ce_out[(size_t)b * P_PAD + p] = ce_neg;   // pad priors write 0 -> clean top-K buffer

    rs[tid] = my_conf; rl[tid] = my_loc; rc[tid] = my_pos;
    __syncthreads();
    for (int s2 = TPB / 2; s2 > 0; s2 >>= 1) {
        if (tid < s2) { rs[tid] += rs[tid + s2]; rl[tid] += rl[tid + s2]; rc[tid] += rc[tid + s2]; }
        __syncthreads();
    }
    if (tid == 0) {
        pconf[blockIdx.x] = rs[0];
        ploc[blockIdx.x]  = rl[0];
        atomicAdd(&n_pos[b], rc[0]);          // integer atomic -> deterministic
    }
}

// Stage 3: exact top-K (K = 3*n_pos) sum of ce_neg per batch.
// All ce >= 0 => uint bit order == float order, top bit never set (31 bits).
//   Phase A: two LDS-histogram passes resolve bits [30:20] and [19:9] (integer
//            LDS atomics -> deterministic).
//   Phase B: 9-step bit-descent on bits [8:0]; each count_ge(t) is computed
//            with V_WMMA_I32_16X16X64_IU8 (A = 1024 packed 0/1 predicate bytes
//            per wave, B = all-ones) -> exact tensorized popcount.
//   Phase C: closed-form sum with exact tie handling:
//            conf_neg = sum(x > v) + (K - count(x > v)) * v.
__global__ void k_topk(const float* __restrict__ ce,
                       const int*   __restrict__ n_pos,
                       double* __restrict__ conf_neg) {
    int b = blockIdx.x, tid = threadIdx.x;
    const unsigned* u = (const unsigned*)(ce + (size_t)b * P_PAD);
    int K = 3 * n_pos[b];

    __shared__ int    hist[2048];
    __shared__ int    s_b0, s_above, s_cnt;
    __shared__ unsigned s_prefix;
    __shared__ double rs[TPB];
    __shared__ int    rc[TPB];

    if (K <= 0) { if (tid == 0) conf_neg[b] = 0.0; return; }   // uniform branch

    unsigned v = 0;
    if (K < P_PAD) {                                           // else v=0: sum-all case
        // ---- Phase A, level 0: bits [30:20] ----
        for (int i = tid; i < 2048; i += TPB) hist[i] = 0;
        __syncthreads();
        for (int i = tid; i < P_PAD; i += TPB) atomicAdd(&hist[u[i] >> 20], 1);
        __syncthreads();
        if (tid == 0) {
            int acc = 0, b0 = 0;
            for (int t = 2047; t >= 0; --t) {
                int h = hist[t];
                if (acc + h >= K) { b0 = t; break; }
                acc += h;
            }
            s_b0 = b0; s_above = acc;
        }
        __syncthreads();
        unsigned b0 = (unsigned)s_b0;
        int above = s_above;
        // ---- Phase A, level 1: bits [19:9] within bucket b0 ----
        __syncthreads();
        for (int i = tid; i < 2048; i += TPB) hist[i] = 0;
        __syncthreads();
        for (int i = tid; i < P_PAD; i += TPB) {
            unsigned x = u[i];
            if ((x >> 20) == b0) atomicAdd(&hist[(x >> 9) & 0x7FF], 1);
        }
        __syncthreads();
        if (tid == 0) {
            int acc = above, b1 = 0;
            for (int t = 2047; t >= 0; --t) {
                int h = hist[t];
                if (acc + h >= K) { b1 = t; break; }
                acc += h;
            }
            s_prefix = ((b0 << 11) | (unsigned)b1) << 9;
        }
        __syncthreads();
        unsigned prefix = s_prefix;

        // ---- Phase B: WMMA bit-descent, bits 8..0 ----
        int wave = tid >> 5, lane = tid & 31;
        v8i bones = {0x01010101, 0x01010101, 0x01010101, 0x01010101,
                     0x01010101, 0x01010101, 0x01010101, 0x01010101};
        for (int bit = 8; bit >= 0; --bit) {
            unsigned t = prefix | (1u << bit);
            if (tid == 0) s_cnt = 0;
            __syncthreads();

            v8i acc = {0, 0, 0, 0, 0, 0, 0, 0};
#pragma unroll
            for (int round = 0; round < 3; ++round) {         // 3 * 8 waves * 1024 = 24576
                int base = round * 8192 + wave * 1024;
                v8i a;
#pragma unroll
                for (int w = 0; w < 8; w++) {
                    unsigned pw = 0;
#pragma unroll
                    for (int jj = 0; jj < 4; jj++) {
                        int j = w * 4 + jj;
                        unsigned x = u[base + j * 32 + lane]; // coalesced b32, L2-hot
                        pw |= (x >= t ? 1u : 0u) << (8 * jj);
                    }
                    a[w] = (int)pw;
                }
                // EXEC all-ones (uniform flow); 16 identical output columns ->
                // sum of all D elements per wave = 16 * (#true predicates).
                acc = __builtin_amdgcn_wmma_i32_16x16x64_iu8(false, a, false, bones,
                                                             acc, false, false);
            }
            int lt = acc[0] + acc[1] + acc[2] + acc[3] + acc[4] + acc[5] + acc[6] + acc[7];
            atomicAdd(&s_cnt, lt);                             // integer -> deterministic
            __syncthreads();
            if ((s_cnt >> 4) >= K) prefix = t;                 // uniform update
            __syncthreads();
        }
        v = prefix;
    }

    // ---- Phase C: exact sum with tie handling ----
    float vf = __uint_as_float(v);
    double sum_gt = 0.0; int cnt_gt = 0;
    for (int i = tid; i < P_PAD; i += TPB) {
        unsigned x = u[i];
        if (x > v) { sum_gt += (double)__uint_as_float(x); cnt_gt++; }
    }
    rs[tid] = sum_gt; rc[tid] = cnt_gt;
    __syncthreads();
    for (int s2 = TPB / 2; s2 > 0; s2 >>= 1) {
        if (tid < s2) { rs[tid] += rs[tid + s2]; rc[tid] += rc[tid + s2]; }
        __syncthreads();
    }
    if (tid == 0) conf_neg[b] = rs[0] + (double)(K - rc[0]) * (double)vf;
}

__global__ void k_final(const double* __restrict__ pconf,
                        const double* __restrict__ ploc,
                        const double* __restrict__ conf_neg,
                        const int*    __restrict__ n_pos,
                        float* __restrict__ out) {
    int tid = threadIdx.x;
    __shared__ double rs[TPB], rl[TPB];
    double c = 0.0, l = 0.0;
    for (int i = tid; i < NBLK2; i += TPB) { c += pconf[i]; l += ploc[i]; }
    rs[tid] = c; rl[tid] = l;
    __syncthreads();
    for (int s2 = TPB / 2; s2 > 0; s2 >>= 1) {
        if (tid < s2) { rs[tid] += rs[tid + s2]; rl[tid] += rl[tid + s2]; }
        __syncthreads();
    }
    if (tid == 0) {
        double cn = 0.0; long np = 0;
        for (int bb = 0; bb < B_SZ; bb++) { cn += conf_neg[bb]; np += n_pos[bb]; }
        double npd = (double)np;
        double conf_loss = (rs[0] + cn) / npd;
        double loc_loss  = 10.0 * rl[0] / (npd * 4.0);
        out[0] = (float)(conf_loss + loc_loss);
        out[1] = (float)conf_loss;
        out[2] = (float)loc_loss;
    }
}

extern "C" void kernel_launch(void* const* d_in, const int* in_sizes, int n_in,
                              void* d_out, int out_size, void* d_ws, size_t ws_size,
                              hipStream_t stream) {
    (void)in_sizes; (void)n_in; (void)out_size; (void)ws_size;
    const float* pred_cls  = (const float*)d_in[0];
    const float* pred_loc  = (const float*)d_in[1];
    const float* gt_boxes  = (const float*)d_in[2];
    const int*   gt_labels = (const int*)  d_in[3];
    const float* anc       = (const float*)d_in[4];
    float* out = (float*)d_out;

    // Workspace layout (all 8-byte aligned, ~3.2 MB total)
    char* ws = (char*)d_ws;
    float*  ce    = (float*) (ws);                 // [32][24576] f32 = 3,145,728 B
    int*    obj   = (int*)   (ws + 3145728);       // [32][24]    i32 =     3,072 B
    int*    npos  = (int*)   (ws + 3148800);       // [32]        i32 (+pad)  128 B
    double* cneg  = (double*)(ws + 3148928);       // [32]        f64 =       256 B
    double* pconf = (double*)(ws + 3149184);       // [3072]      f64 =    24,576 B
    double* ploc  = (double*)(ws + 3173760);       // [3072]      f64 =    24,576 B

    k_init  <<<1, 32, 0, stream>>>(npos);
    k_stage1<<<B_SZ, TPB, 0, stream>>>(gt_boxes, anc, obj);
    k_stage2<<<NBLK2, TPB, 0, stream>>>(pred_cls, pred_loc, gt_boxes, gt_labels,
                                        anc, obj, ce, npos, pconf, ploc);
    k_topk  <<<B_SZ, TPB, 0, stream>>>(ce, npos, cneg);
    k_final <<<1, TPB, 0, stream>>>(pconf, ploc, cneg, npos, out);
}
